// BahdanauAttention_17411797418465
// MI455X (gfx1250) — matrix operand order repaired and checked by reading, not compiled, not run on hardware
//
#include <hip/hip_runtime.h>
#include <hip/hip_bf16.h>
#include <stdint.h>

// ---------------------------------------------------------------------------
// Bahdanau attention, fused for MI455X (gfx1250, wave32, WMMA).
// B=8, T=128, S=256, H=512.
// Strategy: all matmuls in bf16 hi+lo split (3x v_wmma_f32_16x16x32_bf16 per
// K-step ~ fp32 accuracy); split precomputed once so GEMM loop is loads+WMMA.
// Scores (the FLOP hotspot, 134M tanh) use hardware v_tanh_f32.
// ---------------------------------------------------------------------------

#define BB 8
#define TT 128
#define SS 256
#define HH 512

typedef __attribute__((ext_vector_type(16))) __bf16 v16bf;
typedef __attribute__((ext_vector_type(8)))  __bf16 v8bf;
typedef __attribute__((ext_vector_type(4)))  __bf16 v4bf;
typedef __attribute__((ext_vector_type(8)))  float  v8f;

// ----------------------------- helpers -------------------------------------

static __device__ __forceinline__ unsigned short f32_to_bf16_rne(float f) {
  unsigned int u = __builtin_bit_cast(unsigned int, f);
  unsigned int r = u + 0x7FFFu + ((u >> 16) & 1u);   // round-to-nearest-even
  return (unsigned short)(r >> 16);
}
static __device__ __forceinline__ float bf16_to_f32(unsigned short h) {
  unsigned int u = ((unsigned int)h) << 16;
  return __builtin_bit_cast(float, u);
}
static __device__ __forceinline__ __bf16 bfbits(unsigned short h) {
  return __builtin_bit_cast(__bf16, h);
}

// gfx1250 hardware transcendental tanh. ISA: TRANS result needs one
// independent instruction before use -> trailing v_nop inside the asm.
static __device__ __forceinline__ float fast_tanhf(float x) {
  float r;
  asm volatile("v_tanh_f32 %0, %1\n\tv_nop" : "=v"(r) : "v"(x));
  return r;
}

// ---------------------------------------------------------------------------
// Elementwise fp32 -> (bf16 hi, bf16 lo-residual) split. n4 = n/4.
// ---------------------------------------------------------------------------
__global__ __launch_bounds__(256) void split_bf16_kernel(
    const float* __restrict__ x, __bf16* __restrict__ hi,
    __bf16* __restrict__ lo, int n4) {
  int i = blockIdx.x * blockDim.x + threadIdx.x;
  if (i >= n4) return;
  float4 f = ((const float4*)x)[i];
  float v[4] = {f.x, f.y, f.z, f.w};
  v4bf h, l;
#pragma unroll
  for (int j = 0; j < 4; ++j) {
    unsigned short hs = f32_to_bf16_rne(v[j]);
    h[j] = bfbits(hs);
    l[j] = bfbits(f32_to_bf16_rne(v[j] - bf16_to_f32(hs)));
  }
  ((v4bf*)hi)[i] = h;
  ((v4bf*)lo)[i] = l;
}

// ---------------------------------------------------------------------------
// A-matrix 16x32 bf16 fragment: lane = (khalf<<4)|m,
// element j <-> k = khalf*16 + j (sequential, matches B).
// `p` = &A[row_m][kBlock] ; one 32B load, no ALU conversion.
// ---------------------------------------------------------------------------
static __device__ __forceinline__ v16bf load_a_frag(const __bf16* p, int khalf) {
  return *(const v16bf*)(p + (khalf << 4));
}

// B-matrix 32x16 bf16 fragment: lane = (khalf<<4)|n, element j <-> k =
// khalf*16 + j (sequential).  `p` = &W[col_n][kBlock + khalf*16]; 32B load.
static __device__ __forceinline__ v16bf load_b_frag(const __bf16* p) {
  return *(const v16bf*)(p);
}

// ---------------------------------------------------------------------------
// C[m,n] = (tanh?)( sum_k Avirt[m,k] * W[n,k] ),  Avirt = [A0 | A1] split at
// kSplit (lda0 = kSplit, lda1 = K-kSplit), W row-major (N, ldw=K), all bf16
// hi/lo pre-split. One wave computes a 16x64 tile; 3 WMMAs per K-step.
// ---------------------------------------------------------------------------
__global__ __launch_bounds__(256) void wmma_gemm_xwt(
    const __bf16* __restrict__ A0h, const __bf16* __restrict__ A0l,
    const __bf16* __restrict__ A1h, const __bf16* __restrict__ A1l,
    int kSplit,
    const __bf16* __restrict__ Wh, const __bf16* __restrict__ Wl,
    float* __restrict__ C, int M, int N, int K, int applyTanh) {
  const int lane = threadIdx.x & 31;
  const int wid  = blockIdx.x * (blockDim.x >> 5) + (threadIdx.x >> 5);
  const int tilesN = N >> 6;                 // 64-wide column tiles
  const int tilesM = M >> 4;
  if (wid >= tilesM * tilesN) return;        // wave-uniform guard
  const int row0 = (wid / tilesN) << 4;
  const int col0 = (wid % tilesN) << 6;

  const int mA    = row0 + (lane & 15);      // row this lane loads for A
  const int khalf = lane >> 4;
  const int lda1  = K - kSplit;

  v8f acc[4];
#pragma unroll
  for (int c = 0; c < 4; ++c)
    acc[c] = (v8f){0.f, 0.f, 0.f, 0.f, 0.f, 0.f, 0.f, 0.f};

  for (int kb = 0; kb < K; kb += 32) {
    const __bf16 *pah, *pal;
    if (kb < kSplit) {
      size_t off = (size_t)mA * kSplit + kb;
      pah = A0h + off; pal = A0l + off;
    } else {
      size_t off = (size_t)mA * lda1 + (kb - kSplit);
      pah = A1h + off; pal = A1l + off;
    }
    v16bf ah = load_a_frag(pah, khalf);
    v16bf al = load_a_frag(pal, khalf);
#pragma unroll
    for (int c = 0; c < 4; ++c) {
      const size_t boff = (size_t)(col0 + (c << 4) + (lane & 15)) * K + kb +
                          (khalf << 4);
      __builtin_prefetch(Wh + boff + 32, 0, 1);   // next K-block
      v16bf bh = load_b_frag(Wh + boff);
      v16bf bl = load_b_frag(Wl + boff);
      // fp32-ish accuracy: hi*hi + hi*lo + lo*hi
      acc[c] = __builtin_amdgcn_wmma_f32_16x16x32_bf16(
          false, ah, false, bh, (short)0, acc[c], false, false);
      acc[c] = __builtin_amdgcn_wmma_f32_16x16x32_bf16(
          false, ah, false, bl, (short)0, acc[c], false, false);
      acc[c] = __builtin_amdgcn_wmma_f32_16x16x32_bf16(
          false, al, false, bh, (short)0, acc[c], false, false);
    }
  }

  // D layout: lane&15 = n, VGPR r -> m = 8*(lane>=16) + r
  const int nloc  = lane & 15;
  const int mhalf = khalf << 3;
#pragma unroll
  for (int c = 0; c < 4; ++c) {
#pragma unroll
    for (int r = 0; r < 8; ++r) {
      float v = acc[c][r];
      if (applyTanh) v = fast_tanhf(v);
      C[(size_t)(row0 + mhalf + r) * N + col0 + (c << 4) + nloc] = v;
    }
  }
}

// ---------------------------------------------------------------------------
// Fused: scores[b,t,s] = sum_h v[h]*tanh(encp[b,s,h]+qryp[b,t,h]);
// mask s>=len -> -inf; softmax over s; attn_w out; ctx[b,t,h]=sum_s w*enc,
// emitted directly as bf16 hi/lo for the output GEMM.
// One 256-thread block (8 wave32) per (b,t). q/v chunks live in registers.
// ---------------------------------------------------------------------------
__global__ __launch_bounds__(256) void attn_softmax_ctx(
    const float* __restrict__ encp, const float* __restrict__ qryp,
    const float* __restrict__ enc,  const float* __restrict__ vvec,
    const int* __restrict__ src_len,
    float* __restrict__ attn_w,
    __bf16* __restrict__ ctx_h, __bf16* __restrict__ ctx_l) {
  __shared__ float sc[SS];
  __shared__ float red[256];

  const int b    = blockIdx.x >> 7;          // T = 128
  const int t    = blockIdx.x & 127;
  const int tid  = threadIdx.x;
  const int lane = tid & 31;
  const int wave = tid >> 5;

  // Per-lane register copies of q-proj row and v (16 contiguous fp32 each).
  float qreg[16], vreg[16];
  {
    const float* qp = qryp + (size_t)(b * TT + t) * HH + lane * 16;
    const float* vp = vvec + lane * 16;
#pragma unroll
    for (int j4 = 0; j4 < 4; ++j4) {
      float4 q = *(const float4*)(qp + j4 * 4);
      float4 v = *(const float4*)(vp + j4 * 4);
      qreg[j4 * 4 + 0] = q.x; qreg[j4 * 4 + 1] = q.y;
      qreg[j4 * 4 + 2] = q.z; qreg[j4 * 4 + 3] = q.w;
      vreg[j4 * 4 + 0] = v.x; vreg[j4 * 4 + 1] = v.y;
      vreg[j4 * 4 + 2] = v.z; vreg[j4 * 4 + 3] = v.w;
    }
  }

  // ---- scores: each wave covers s = wave, wave+8, ... (32 rows) ----
  for (int s = wave; s < SS; s += 8) {
    const float* ep = encp + (size_t)(b * SS + s) * HH + lane * 16;
    float p = 0.f;
#pragma unroll
    for (int j4 = 0; j4 < 4; ++j4) {
      float4 e = *(const float4*)(ep + j4 * 4);
      p += vreg[j4 * 4 + 0] * fast_tanhf(e.x + qreg[j4 * 4 + 0]);
      p += vreg[j4 * 4 + 1] * fast_tanhf(e.y + qreg[j4 * 4 + 1]);
      p += vreg[j4 * 4 + 2] * fast_tanhf(e.z + qreg[j4 * 4 + 2]);
      p += vreg[j4 * 4 + 3] * fast_tanhf(e.w + qreg[j4 * 4 + 3]);
    }
#pragma unroll
    for (int off = 16; off > 0; off >>= 1) p += __shfl_xor(p, off, 32);
    if (lane == 0) sc[s] = p;
  }
  __syncthreads();

  // ---- mask + softmax over s (tid == s, S == 256) ----
  const int len = src_len[b];
  const float NEG_INF = -__builtin_inff();
  float s0 = (tid < len) ? sc[tid] : NEG_INF;

  red[tid] = s0;
  __syncthreads();
#pragma unroll
  for (int off = 128; off > 0; off >>= 1) {
    if (tid < off) red[tid] = fmaxf(red[tid], red[tid + off]);
    __syncthreads();
  }
  const float mx = red[0];
  __syncthreads();

  float e = (tid < len) ? __expf(s0 - mx) : 0.f;
  red[tid] = e;
  __syncthreads();
#pragma unroll
  for (int off = 128; off > 0; off >>= 1) {
    if (tid < off) red[tid] += red[tid + off];
    __syncthreads();
  }
  const float inv = 1.f / red[0];
  __syncthreads();

  const float w = e * inv;
  attn_w[(size_t)(b * TT + t) * SS + tid] = w;
  sc[tid] = w;                                // reuse as weight buffer
  __syncthreads();

  // ---- context: ctx[b,t,h] = sum_s w[s] * enc[b,s,h] (coalesced in h),
  //      stored pre-split as bf16 hi/lo for the output WMMA GEMM.
  const float* eb = enc + (size_t)(b * SS) * HH;
#pragma unroll
  for (int hh = 0; hh < HH; hh += 256) {
    const int h = hh + tid;
    float a = 0.f;
#pragma unroll 4
    for (int s = 0; s < SS; ++s) a += sc[s] * eb[(size_t)s * HH + h];
    const size_t idx = (size_t)(b * TT + t) * HH + h;
    unsigned short hs = f32_to_bf16_rne(a);
    ctx_h[idx] = bfbits(hs);
    ctx_l[idx] = bfbits(f32_to_bf16_rne(a - bf16_to_f32(hs)));
  }
}

// ---------------------------------------------------------------------------

extern "C" void kernel_launch(void* const* d_in, const int* in_sizes, int n_in,
                              void* d_out, int out_size, void* d_ws, size_t ws_size,
                              hipStream_t stream) {
  (void)in_sizes; (void)n_in; (void)out_size; (void)ws_size;

  const float* query = (const float*)d_in[0];   // (B,T,H)
  const float* enc   = (const float*)d_in[1];   // (B,S,H)
  const int*   slen  = (const int*)d_in[2];     // (B,)
  const float* W_h   = (const float*)d_in[3];   // (H,H)
  const float* W_s   = (const float*)d_in[4];   // (H,H)
  const float* vvec  = (const float*)d_in[5];   // (1,H)
  const float* W_out = (const float*)d_in[6];   // (H,2H)

  const size_t nEnc = (size_t)BB * SS * HH;     // 1,048,576
  const size_t nQry = (size_t)BB * TT * HH;     //   524,288
  const size_t nWh  = (size_t)HH * HH;          //   262,144
  const size_t nWo  = (size_t)HH * 2 * HH;      //   524,288

  // ---- workspace carve (fp32 region, then bf16 region) ----
  float* encp = (float*)d_ws;                   // (B*S, H)
  float* qryp = encp + nEnc;                    // (B*T, H)
  __bf16* bp    = (__bf16*)(qryp + nQry);
  __bf16* enc_h = bp;            bp += nEnc;
  __bf16* enc_l = bp;            bp += nEnc;
  __bf16* qry_h = bp;            bp += nQry;
  __bf16* qry_l = bp;            bp += nQry;
  __bf16* Wh_h  = bp;            bp += nWh;
  __bf16* Wh_l  = bp;            bp += nWh;
  __bf16* Ws_h  = bp;            bp += nWh;
  __bf16* Ws_l  = bp;            bp += nWh;
  __bf16* Wo_h  = bp;            bp += nWo;
  __bf16* Wo_l  = bp;            bp += nWo;
  __bf16* ctx_h = bp;            bp += nQry;
  __bf16* ctx_l = bp;            bp += nQry;

  float* attn_out = (float*)d_out;                    // (B,T,H)
  float* attn_w   = attn_out + nQry;                  // (B,T,S)

  // ---- 0) one-time bf16 hi/lo splits (O(N), bandwidth-trivial) ----
  split_bf16_kernel<<<(int)(nEnc / 4 / 256), 256, 0, stream>>>(enc, enc_h, enc_l, (int)(nEnc / 4));
  split_bf16_kernel<<<(int)(nQry / 4 / 256), 256, 0, stream>>>(query, qry_h, qry_l, (int)(nQry / 4));
  split_bf16_kernel<<<(int)(nWh  / 4 / 256), 256, 0, stream>>>(W_h, Wh_h, Wh_l, (int)(nWh / 4));
  split_bf16_kernel<<<(int)(nWh  / 4 / 256), 256, 0, stream>>>(W_s, Ws_h, Ws_l, (int)(nWh / 4));
  split_bf16_kernel<<<(int)(nWo  / 4 / 256), 256, 0, stream>>>(W_out, Wo_h, Wo_l, (int)(nWo / 4));

  // ---- 1) enc_proj = enc @ W_h^T : M=2048,N=512,K=512 -> 1024 waves ----
  wmma_gemm_xwt<<<128, 256, 0, stream>>>(enc_h, enc_l, enc_h, enc_l, HH,
                                         Wh_h, Wh_l, encp,
                                         BB * SS, HH, HH, 0);
  // ---- 2) qry_proj = query @ W_s^T : M=1024 ----
  wmma_gemm_xwt<<<64, 256, 0, stream>>>(qry_h, qry_l, qry_h, qry_l, HH,
                                        Ws_h, Ws_l, qryp,
                                        BB * TT, HH, HH, 0);
  // ---- 3) fused scores -> mask -> softmax -> context (ctx as bf16 hi/lo) ----
  attn_softmax_ctx<<<BB * TT, 256, 0, stream>>>(encp, qryp, enc, vvec, slen,
                                                attn_w, ctx_h, ctx_l);
  // ---- 4) attn_out = tanh([ctx|query] @ W_out^T) : M=1024,N=512,K=1024 ----
  wmma_gemm_xwt<<<64, 256, 0, stream>>>(ctx_h, ctx_l, qry_h, qry_l, HH,
                                        Wo_h, Wo_l, attn_out,
                                        BB * TT, HH, 2 * HH, 1);
}
